// TokenModel_48000554500161
// MI455X (gfx1250) — compile-verified, matrix operand order checked by reading, not run on hardware
//
#include <hip/hip_runtime.h>
#include <hip/hip_bf16.h>
#include <math.h>

typedef _Float16 v16h __attribute__((ext_vector_type(16)));
typedef float    v8f  __attribute__((ext_vector_type(8)));
typedef int      vi4  __attribute__((vector_size(16)));   // int __vector(4)

#define Bn    64
#define Sn    512
#define Hn    768
#define ATTH  100
#define CLSH  300
#define NTILE 7              // ceil(100/16) column tiles, padded N = 112
#define KSTEPS 24            // 768 / 32
#define CHUNKS 4
#define STEPS_PER_CHUNK 6    // KSTEPS / CHUNKS
#define FRAGS_PER_CHUNK (STEPS_PER_CHUNK * NTILE)   // 42 fragments
#define CHUNK_HALFS (FRAGS_PER_CHUNK * 32 * 16)     // 21504 halfs
#define CHUNK_BYTES (CHUNK_HALFS * 2)               // 43008 bytes

#if __has_builtin(__builtin_amdgcn_global_load_async_to_lds_b128)
#define HAVE_ASYNC_LDS 1
typedef vi4 __attribute__((address_space(1))) gvi4;   // global int4 vector
typedef vi4 __attribute__((address_space(3))) svi4;   // LDS int4 vector
#else
#define HAVE_ASYNC_LDS 0
#endif

// ---------------------------------------------------------------------------
// Kernel 1: pack W1 (768x100 f32) into f16 WMMA B-fragments, padded to N=112.
// Chunk-contiguous layout: fragment f = step*NTILE + t; element e of lane l of
// fragment f lives at half index (f*32 + l)*16 + e, with
//   col = t*16 + (l&15)
//   k   = step*32 + (l>=16 ? 8 : 0) + (e<8 ? e : e+8)
// matching the 16-bit 16x32 WMMA operand layout (mirrored for B).
// ---------------------------------------------------------------------------
__global__ void __launch_bounds__(256) pack_w1(const float* __restrict__ W1,
                                               _Float16* __restrict__ w1p) {
  int idx = blockIdx.x * 256 + threadIdx.x;
  const int total = NTILE * KSTEPS * 32 * 16;
  if (idx >= total) return;
  int e    = idx & 15;
  int lane = (idx >> 4) & 31;
  int f    = idx >> 9;
  int t    = f % NTILE;
  int step = f / NTILE;
  int col  = t * 16 + (lane & 15);
  int kb   = step * 32 + ((lane >> 4) ? 8 : 0);
  int k    = kb + (e < 8 ? e : e + 8);
  float v  = (col < ATTH) ? W1[k * ATTH + col] : 0.0f;
  w1p[idx] = (_Float16)v;
}

// ---------------------------------------------------------------------------
// Kernel 2: token attention head via WMMA.
//   att = sigmoid(tanh(h @ W1 + b1) @ W2 + b2)  for all 32768 tokens.
// One wave per 16-row tile; B fragments staged through LDS per workgroup
// (8 waves share) via gfx1250 async global->LDS copies, cutting L2 B-traffic
// 8x. Also streams the hidden pass-through copy.
// ---------------------------------------------------------------------------
__global__ void __launch_bounds__(256) token_att_gemm(
    const float* __restrict__ hidden, const _Float16* __restrict__ w1p,
    const float* __restrict__ b1, const float* __restrict__ W2,
    const float* __restrict__ b2, float* __restrict__ att_out,
    float* __restrict__ out_hidden)
{
  __shared__ __align__(32) _Float16 w1s[CHUNK_HALFS];   // 42 KB staging buffer

  int wave    = (blockIdx.x * blockDim.x + threadIdx.x) >> 5;
  int lane    = threadIdx.x & 31;
  int lrow    = lane & 15;
  int hi      = lane >> 4;
  int rowBase = wave * 16;
  int row     = rowBase + lrow;

  // C init = broadcast bias per column (same for every row of the tile)
  v8f acc[NTILE];
  for (int t = 0; t < NTILE; ++t) {
    int n = t * 16 + lrow;
    float bv = (n < ATTH) ? b1[n] : 0.0f;
    for (int i = 0; i < 8; ++i) acc[t][i] = bv;
  }

  const v16h* bs = (const v16h*)w1s;

  for (int c = 0; c < CHUNKS; ++c) {
    // ---- stage chunk c of packed W1 into LDS (42 KB, 2688 x 16B units) ----
    const char* gsrc = (const char*)w1p + (size_t)c * CHUNK_BYTES;
#if HAVE_ASYNC_LDS
    for (int u = threadIdx.x; u < CHUNK_BYTES / 16; u += 256) {
      __builtin_amdgcn_global_load_async_to_lds_b128(
          (gvi4*)((char*)gsrc + (size_t)u * 16),
          (svi4*)((char*)w1s + (size_t)u * 16),
          0, 0);
    }
    asm volatile("s_wait_asynccnt 0" ::: "memory");
#else
    for (int u = threadIdx.x; u < CHUNK_BYTES / 16; u += 256) {
      ((int4*)w1s)[u] = ((const int4*)gsrc)[u];
    }
#endif
    __syncthreads();

    // ---- 6 K-steps from this chunk ----
    for (int i = 0; i < STEPS_PER_CHUNK; ++i) {
      int step = c * STEPS_PER_CHUNK + i;
      int kb = step * 32 + hi * 8;             // 32B-aligned float offset
      const float* ap = hidden + (size_t)row * Hn + kb;
      __builtin_prefetch(ap + 32, 0, 0);       // global_prefetch_b8 (next step)
      float4 a0 = ((const float4*)ap)[0];
      float4 a1 = ((const float4*)ap)[1];
      float4 a2 = ((const float4*)(ap + 16))[0];
      float4 a3 = ((const float4*)(ap + 16))[1];

      // fused pass-through copy of hidden_states into d_out
      float* op = out_hidden + (size_t)row * Hn + kb;
      op[0]  = a0.x; op[1]  = a0.y; op[2]  = a0.z; op[3]  = a0.w;
      op[4]  = a1.x; op[5]  = a1.y; op[6]  = a1.z; op[7]  = a1.w;
      op[16] = a2.x; op[17] = a2.y; op[18] = a2.z; op[19] = a2.w;
      op[20] = a3.x; op[21] = a3.y; op[22] = a3.z; op[23] = a3.w;

      v16h A;
      A[0]  = (_Float16)a0.x; A[1]  = (_Float16)a0.y;
      A[2]  = (_Float16)a0.z; A[3]  = (_Float16)a0.w;
      A[4]  = (_Float16)a1.x; A[5]  = (_Float16)a1.y;
      A[6]  = (_Float16)a1.z; A[7]  = (_Float16)a1.w;
      A[8]  = (_Float16)a2.x; A[9]  = (_Float16)a2.y;
      A[10] = (_Float16)a2.z; A[11] = (_Float16)a2.w;
      A[12] = (_Float16)a3.x; A[13] = (_Float16)a3.y;
      A[14] = (_Float16)a3.z; A[15] = (_Float16)a3.w;

      for (int t = 0; t < NTILE; ++t) {
        v16h Bf = bs[(i * NTILE + t) * 32 + lane];   // ds_load from LDS
        acc[t] = __builtin_amdgcn_wmma_f32_16x16x32_f16(
            false, A, false, Bf, (short)0, acc[t], false, false);
      }
    }
    __syncthreads();   // before next chunk overwrites the staging buffer
  }

  // tanh then N->1 projection with W2. C layout: lane holds col n=t*16+lrow,
  // VGPR r holds row M=r (lanes 0-15) / M=r+8 (lanes 16-31).
  float partial[8];
  for (int r = 0; r < 8; ++r) partial[r] = 0.0f;
  for (int t = 0; t < NTILE; ++t) {
    int n = t * 16 + lrow;
    if (n < ATTH) {
      float w2v = W2[n];
      for (int r = 0; r < 8; ++r) partial[r] += tanhf(acc[t][r]) * w2v;
    }
  }
  float bb = b2[0];
  for (int r = 0; r < 8; ++r) {
    float s = partial[r];
    s += __shfl_xor(s, 1);
    s += __shfl_xor(s, 2);
    s += __shfl_xor(s, 4);
    s += __shfl_xor(s, 8);                     // sum within 16-lane half
    if (lrow == 0) {
      float att = 1.0f / (1.0f + expf(-(s + bb)));
      att_out[rowBase + hi * 8 + r] = att;
    }
  }
}

// --------------------------- block reductions ------------------------------
__device__ __forceinline__ float blockReduceAdd(float v, float* red, int tid) {
  red[tid] = v; __syncthreads();
  for (int s = Sn / 2; s > 0; s >>= 1) {
    if (tid < s) red[tid] += red[tid + s];
    __syncthreads();
  }
  float r = red[0]; __syncthreads(); return r;
}
__device__ __forceinline__ float blockReduceMax(float v, float* red, int tid) {
  red[tid] = v; __syncthreads();
  for (int s = Sn / 2; s > 0; s >>= 1) {
    if (tid < s) red[tid] = fmaxf(red[tid], red[tid + s]);
    __syncthreads();
  }
  float r = red[0]; __syncthreads(); return r;
}
__device__ __forceinline__ float blockReduceMin(float v, float* red, int tid) {
  red[tid] = v; __syncthreads();
  for (int s = Sn / 2; s > 0; s >>= 1) {
    if (tid < s) red[tid] = fminf(red[tid], red[tid + s]);
    __syncthreads();
  }
  float r = red[0]; __syncthreads(); return r;
}

// ---------------------------------------------------------------------------
// Kernel 3: per-batch-row segment-max, mask, pooling, classifier, loss parts.
// One workgroup (512 threads) per batch element.
// ---------------------------------------------------------------------------
__global__ void __launch_bounds__(512) row_pool_cls(
    const float* __restrict__ hidden, const int* __restrict__ om,
    const float* __restrict__ labels, const float* __restrict__ att_ws,
    const float* __restrict__ W3, const float* __restrict__ b3,
    const float* __restrict__ W4, const float* __restrict__ b4,
    float* __restrict__ out_masked, float* __restrict__ out_sent,
    float* __restrict__ part)
{
  __shared__ float attS[Sn];
  __shared__ float labS[Sn];
  __shared__ float mS[Sn];
  __shared__ float red[Sn];
  __shared__ int   contS[Sn];
  __shared__ float pooled[Hn];
  __shared__ float cls[CLSH];

  int b = blockIdx.x, tid = threadIdx.x;
  attS[tid]  = att_ws[b * Sn + tid];
  contS[tid] = (om[(b * Sn + tid) * 2] != 0) ? 1 : 0;
  labS[tid]  = labels[b * Sn + tid];
  __syncthreads();

  // segment max scattered onto word roots; continuations masked to zero
  float m = 0.0f;
  if (!contS[tid]) {
    float v = attS[tid];
    for (int j = tid + 1; j < Sn && contS[j]; ++j) v = fmaxf(v, attS[j]);
    if (labS[tid] != -1.0f) m = v;
  }
  mS[tid] = m;
  out_masked[b * Sn + tid] = m;
  __syncthreads();

  float sumP = blockReduceAdd(m, red, tid);              // beta=1 -> p=masked
  float maxM = blockReduceMax(m, red, tid);
  float ones = (m == 0.0f) ? 1.0f : m;
  float minO = blockReduceMin(ones, red, tid);
  float maxL = blockReduceMax(labS[tid], red, tid);
  float zl   = (labS[tid] == 1.0f) ? 1.0f : 0.0f;
  float tok  = blockReduceAdd((m - zl) * (m - zl), red, tid);

  // soft-attention pooling: pooled[h] = sum_s hidden[b,s,h]*masked[s]/sumP
  float invS = 1.0f / sumP;
  for (int h = tid; h < Hn; h += Sn) {
    const float* hp = hidden + (size_t)b * Sn * Hn + h;
    float a = 0.0f;
    for (int s = 0; s < Sn; ++s) a += hp[(size_t)s * Hn] * mS[s];
    pooled[h] = a * invS;
  }
  __syncthreads();

  // classifier head
  if (tid < CLSH) {
    float z = b3[tid];
    for (int h = 0; h < Hn; ++h) z += pooled[h] * W3[h * CLSH + tid];
    cls[tid] = tanhf(z);
  }
  __syncthreads();
  float c = (tid < CLSH) ? cls[tid] * W4[tid] : 0.0f;
  float z4 = blockReduceAdd(c, red, tid);

  if (tid == 0) {
    float sent = 1.0f / (1.0f + expf(-(z4 + b4[0])));
    out_sent[b] = sent;
    float ds = sent - maxL;
    float dr = maxM - maxL;
    part[b * 4 + 0] = ds * ds;   // sentence loss part
    part[b * 4 + 1] = tok;       // token loss part
    part[b * 4 + 2] = minO * minO;
    part[b * 4 + 3] = dr * dr;
  }
}

// ---------------------------------------------------------------------------
// Kernel 4: deterministic final loss reduction (fixed order, no atomics).
// ---------------------------------------------------------------------------
__global__ void finalize(const float* __restrict__ part, float* __restrict__ out) {
  if (blockIdx.x == 0 && threadIdx.x == 0) {
    float sl = 0.f, tl = 0.f, ra = 0.f, rb = 0.f;
    for (int b = 0; b < Bn; ++b) {
      sl += part[b * 4 + 0];
      tl += part[b * 4 + 1];
      ra += part[b * 4 + 2];
      rb += part[b * 4 + 3];
    }
    out[1] = sl; out[2] = tl; out[3] = ra; out[4] = rb;
    out[0] = sl + tl + 0.01f * (ra + rb);
  }
}

// ---------------------------------------------------------------------------
extern "C" void kernel_launch(void* const* d_in, const int* in_sizes, int n_in,
                              void* d_out, int out_size, void* d_ws, size_t ws_size,
                              hipStream_t stream) {
  const float* hidden = (const float*)d_in[0];
  const int*   om     = (const int*)  d_in[1];
  const float* labels = (const float*)d_in[2];
  const float* W1     = (const float*)d_in[3];
  const float* b1     = (const float*)d_in[4];
  const float* W2     = (const float*)d_in[5];
  const float* b2     = (const float*)d_in[6];
  const float* W3     = (const float*)d_in[7];
  const float* b3     = (const float*)d_in[8];
  const float* W4     = (const float*)d_in[9];
  const float* b4     = (const float*)d_in[10];

  float* out = (float*)d_out;
  const size_t HID = (size_t)Bn * Sn * Hn;            // 25,165,824
  float* outHidden = out + 5;
  float* outMasked = out + 5 + HID;
  float* outSent   = out + 5 + HID + (size_t)Bn * Sn;

  // workspace layout (all offsets 32B aligned)
  char* ws = (char*)d_ws;
  float*    att_ws = (float*)ws;                                  // 32768 f32
  _Float16* w1p    = (_Float16*)(ws + (size_t)Bn * Sn * 4);       // 86016 f16
  float*    part   = (float*)(ws + (size_t)Bn * Sn * 4 + (size_t)NTILE * KSTEPS * 32 * 16 * 2);

  // 1) pack W1 -> f16 WMMA fragments (86016 elems)
  pack_w1<<<(NTILE * KSTEPS * 32 * 16 + 255) / 256, 256, 0, stream>>>(W1, w1p);

  // 2) token attention GEMM via WMMA: 2048 waves = 256 blocks x 8 waves,
  //    LDS-staged B fragments, fused hidden_states pass-through output.
  token_att_gemm<<<256, 256, 0, stream>>>(hidden, w1p, b1, W2, b2,
                                          att_ws, outHidden);

  // 3) per-batch segment-max / pooling / classifier / loss partials
  row_pool_cls<<<Bn, Sn, 0, stream>>>(hidden, om, labels, att_ws,
                                      W3, b3, W4, b4,
                                      outMasked, outSent, part);

  // 4) deterministic scalar loss reduction
  finalize<<<1, 32, 0, stream>>>(part, out);
}
